// QuantizePowerConv2d_53395033424602
// MI455X (gfx1250) — 8-bit matrix operand order repaired and checked by reading, compiled, not run on hardware
//
#include <hip/hip_runtime.h>
#include <hip/hip_bf16.h>

typedef __attribute__((ext_vector_type(8)))  int   v8i;
typedef __attribute__((ext_vector_type(16))) int   v16i;
typedef __attribute__((ext_vector_type(8)))  float v8f;

#define NB   32
#define CIN  64
#define KOUT 64
#define HW   56
#define PIX  (HW*HW)
#define NPIX (NB*PIX)
#define NW   36864
#define BPAD (640*64)
#define NTOT (NB*KOUT*PIX)
#define NTILES (NPIX/16)
#define WAVES 4
#define BBYTES (NW + 3*BPAD)


__global__ __launch_bounds__(512)
void qpc_prep_weights(const float* __restrict__ w, float* __restrict__ ws_sw,
                      signed char* __restrict__ Bmain, unsigned char* __restrict__ BbitF,
                      float* __restrict__ accums) {
    __shared__ float red[512];
    const int tid = threadIdx.x;
    float m = 0.f;
    for (int i = tid; i < NW; i += 512) m = fmaxf(m, fabsf(w[i]));
    red[tid] = m;
    __syncthreads();
    for (int s = 256; s > 0; s >>= 1) {
        if (tid < s) red[tid] = fmaxf(red[tid], red[tid + s]);
        __syncthreads();
    }
    const float sw = red[0];
    if (tid == 0) ws_sw[0] = sw;
    if (tid < 24) accums[tid] = 0.f;
    for (int i = tid; i < 3 * (BPAD - NW); i += 512) {
        const int j = i / (BPAD - NW);
        const int o = i - j * (BPAD - NW);
        BbitF[j * BPAD + NW + o] = 0;
    }
    for (int i = tid; i < NW; i += 512) {
        const int K = i / 576;
        const int c = (i / 9) % 64;
        const int t = i % 9;
        const int sB = (c & 3) | (((c >> 2) & 1) << 2) | (((c >> 4) & 1) << 3) | (((c >> 5) & 1) << 4);
        const int pos = ((t * 64 + ((c >> 3) & 1) * 16 + (K & 15) + 32 * (sB >> 4)) << 6) + (K >> 4) * 16 + (sB & 15);
        const float q = rintf(fminf(fmaxf(w[i] / sw, -1.f), 1.f) * 7.f);
        Bmain[pos] = (signed char)(int)q;
        const int aw = (int)fabsf(q);
        #pragma unroll
        for (int j = 0; j < 3; ++j)
            BbitF[j * BPAD + pos] = ((aw >> j) & 1) ? 0x4A : 0x38;
    }
}

__global__ __launch_bounds__(256)
void qpc_quant_input(const float* __restrict__ x, const float* __restrict__ s_in_p,
                     unsigned char* __restrict__ aplane) {
    const int tid = blockIdx.x * 256 + threadIdx.x;
    if (tid >= NTOT) return;
    const float inv = 1.f / s_in_p[0];
    const int n  = tid / (CIN * PIX);
    const int c  = (tid / PIX) % CIN;
    const int hw = tid % PIX;
    float a = rintf(fminf(fmaxf(x[tid] * inv, -1.f), 1.f) * 7.f);
    int ai = (int)a; if (ai < 0) ai = 0;
    const unsigned bits = (0x75316420u >> (ai << 2)) & 7u;
    aplane[((n * PIX + hw) << 6) + c] = (unsigned char)(ai | (bits << 4));
}

__global__ __launch_bounds__(128)
void qpc_gemm(const unsigned char* __restrict__ aplane,
              const unsigned char* __restrict__ Bmain,
              const unsigned char* __restrict__ BbitF,
              const float* __restrict__ bias,
              const float* __restrict__ s_in_p, const float* __restrict__ s_out_p,
              const float* __restrict__ ws_sw,
              float* __restrict__ sums, float* __restrict__ sumsq,
              float* __restrict__ qout) {
    __shared__ __align__(16) unsigned ldsu[WAVES * 2560];

    const int lane = threadIdx.x & 31;
    const int wid  = threadIdx.x >> 5;
    const int tile = blockIdx.x * WAVES + wid;
    const int basePix = tile * 16;
    unsigned* myLds = ldsu + wid * 2560;
    const int mrow = lane & 15;
    const int half = lane >> 4;

    for (int i = threadIdx.x * 128; i < BBYTES; i += 128 * 128)
        __builtin_prefetch(Bmain + i, 0, 1);

    for (int i = lane; i < 640; i += 32) {
        const int row = i / 40;
        const int rem = i - row * 40;
        const int tap = rem >> 2;
        const int d4  = rem & 3;
        const int p = basePix + row;
        const int n = p / PIX, hw = p - n * PIX;
        const int h = hw / HW, wq = hw - h * HW;
        const int hh = h + tap / 3 - 1, ww = wq + tap % 3 - 1;
        uint4 val = make_uint4(0u, 0u, 0u, 0u);
        if (tap < 9 && hh >= 0 && hh < HW && ww >= 0 && ww < HW)
            val = *(const uint4*)(aplane + (((n * HW + hh) * HW + ww) << 6) + (d4 << 4));
        *(uint4*)(myLds + row * 160 + tap * 16 + d4 * 4) = val;
    }
    __syncthreads();

    const unsigned* rowp = myLds + mrow * 160;
    auto loadRaw = [&](int t) {
        v8i a;
        #pragma unroll
        for (int v = 0; v < 8; ++v) {
            const int off = half * 8 + (v & 1) * 4 + ((v >> 1) & 1) * 16 + (v >> 2) * 32;
            a[v] = (int)rowp[t * 16 + (off >> 2)];
        }
        return a;
    };
    auto loadB64 = [&](const unsigned char* B, int t, int nt) {
        const int kb = half * 16 + mrow;
        const uint4 x0 = *(const uint4*)(B + ((t * 64 + kb)      << 6) + nt * 16);
        const uint4 x1 = *(const uint4*)(B + ((t * 64 + kb + 32) << 6) + nt * 16);
        v8i b;
        b[0] = (int)x0.x; b[1] = (int)x0.y; b[2] = (int)x0.z; b[3] = (int)x0.w;
        b[4] = (int)x1.x; b[5] = (int)x1.y; b[6] = (int)x1.z; b[7] = (int)x1.w;
        return b;
    };

    v8i accM[4];
    #pragma unroll
    for (int nt = 0; nt < 4; ++nt) accM[nt] = (v8i)0;
    #pragma unroll
    for (int t = 0; t < 9; ++t) {
        const v8i wrd = loadRaw(t);
        v8i a;
        #pragma unroll
        for (int v = 0; v < 8; ++v) a[v] = wrd[v] & 0x07070707;
        #pragma unroll
        for (int nt = 0; nt < 4; ++nt) {
            const v8i b = loadB64(Bmain, t, nt);
            accM[nt] = __builtin_amdgcn_wmma_i32_16x16x64_iu8(
                false, a, true, b, accM[nt], false, false);
        }
    }
    {
        const float s_in = s_in_p[0], s_out = s_out_p[0], sw = ws_sw[0];
        const float scale = s_in * sw * (1.f / 49.f);
        const float dq = s_out * (1.f / 127.f);
        #pragma unroll
        for (int nt = 0; nt < 4; ++nt) {
            const int k = nt * 16 + mrow;
            const float bv = bias[k];
            #pragma unroll
            for (int r = 0; r < 8; ++r) {
                const int p = basePix + r + 8 * half;
                const int n = p / PIX, hw = p - n * PIX;
                const float o = (float)accM[nt][r] * scale + bv;
                const float q = rintf(fminf(fmaxf(o / s_out, -1.f), 1.f) * 127.f) * dq;
                qout[(n * KOUT + k) * PIX + hw] = q;
            }
        }
    }

    auto loadAbit = [&](int kk, int i) {
        v16i a;
        #pragma unroll
        for (int v = 0; v < 8; ++v) {
            const int off = half * 8 + (v & 1) * 4 + ((v >> 1) & 1) * 16 + (v >> 2) * 32;
            const unsigned w0 = rowp[(2 * kk)     * 16 + (off >> 2)];
            const unsigned w1 = rowp[(2 * kk + 1) * 16 + (off >> 2)];
            a[v]     = (int)(((w0 >> (4 + i)) & 0x01010101u) * 0x38u);
            a[v + 8] = (int)(((w1 >> (4 + i)) & 0x01010101u) * 0x38u);
        }
        return a;
    };
    auto loadB128 = [&](const unsigned char* B, int kk, int nt) {
        v16i b;
        const int kbase = kk * 128 + half * 16 + mrow;
        #pragma unroll
        for (int q = 0; q < 4; ++q) {
            const uint4 x = *(const uint4*)(B + ((kbase + 32 * q) << 6) + nt * 16);
            b[4*q+0] = (int)x.x; b[4*q+1] = (int)x.y;
            b[4*q+2] = (int)x.z; b[4*q+3] = (int)x.w;
        }
        return b;
    };

    #pragma unroll 1
    for (int i = 0; i < 3; ++i) {
        #pragma unroll 1
        for (int j = 0; j < 3; ++j) {
            const unsigned char* Bj = BbitF + j * BPAD;
            v8f acc[4];
            #pragma unroll
            for (int nt = 0; nt < 4; ++nt) acc[nt] = (v8f)0.f;
            #pragma unroll 1
            for (int kk = 0; kk < 5; ++kk) {
                const v16i a = loadAbit(kk, i);
                #pragma unroll
                for (int nt = 0; nt < 4; ++nt) {
                    const v16i b = loadB128(Bj, kk, nt);
                    acc[nt] = __builtin_amdgcn_wmma_f32_16x16x128_fp8_fp8(
                        a, b, (short)0, acc[nt], false, false);
                }
            }
            float s = 0.f, s2 = 0.f;
            #pragma unroll
            for (int nt = 0; nt < 4; ++nt)
                #pragma unroll
                for (int r = 0; r < 8; ++r) {
                    const float f = acc[nt][r];
                    s += f; s2 += f * f;
                }
            #pragma unroll
            for (int o = 16; o > 0; o >>= 1) {
                s  += __shfl_xor(s,  o, 32);
                s2 += __shfl_xor(s2, o, 32);
            }
            if (lane == 0) {
                atomicAdd(&sums [i * 3 + j], s);
                atomicAdd(&sumsq[i * 3 + j], s2);
            }
        }
    }
}

__global__ void qpc_finalize(const float* __restrict__ power_in,
                             const float* __restrict__ sums,
                             const float* __restrict__ sumsq,
                             float* __restrict__ out_p) {
    const float N = (float)NTOT;
    float p = power_in[0];
    for (int k = 0; k < 12; ++k) {
        const int idx = (k < 9) ? k : (k - 9);
        const float s = sums[idx], s2 = sumsq[idx];
        const float mean = s / N;
        const float var = (s2 - N * mean * mean) / (N - 1.f);
        const float RR = sqrtf(fmaxf(var, 0.f)) * 0.1f;
        p += s * (RR + 1.f);
    }
    out_p[0] = p;
}

extern "C" void kernel_launch(void* const* d_in, const int* in_sizes, int n_in,
                              void* d_out, int out_size, void* d_ws, size_t ws_size,
                              hipStream_t stream) {
    const float* x      = (const float*)d_in[0];
    const float* weight = (const float*)d_in[1];
    const float* bias   = (const float*)d_in[2];
    const float* lv_in  = (const float*)d_in[3];
    const float* lv_out = (const float*)d_in[4];
    const float* power  = (const float*)d_in[5];

    char* ws = (char*)d_ws;
    float*         ws_sw   = (float*)ws;
    float*         sums    = (float*)(ws + 16);
    float*         sumsq   = (float*)(ws + 64);
    signed char*   Bmain   = (signed char*)(ws + 256);
    unsigned char* BbitF   = (unsigned char*)(ws + 37120);
    unsigned char* aplane  = (unsigned char*)(ws + 160000);

    float* qout = (float*)d_out;
    float* outp = (float*)d_out + (out_size - 1);

    qpc_prep_weights<<<1, 512, 0, stream>>>(weight, ws_sw, Bmain, BbitF, sums);
    qpc_quant_input<<<(NTOT + 255) / 256, 256, 0, stream>>>(x, lv_in, aplane);
    qpc_gemm<<<NTILES / WAVES, WAVES * 32, 0, stream>>>(
        aplane, (const unsigned char*)Bmain, BbitF, bias, lv_in, lv_out,
        ws_sw, sums, sumsq, qout);
    qpc_finalize<<<1, 1, 0, stream>>>(power, sums, sumsq, outp);
}
